// RNNSpatialRegression_14748917694468
// MI455X (gfx1250) — compile-verified, matrix operand order checked by reading, not run on hardware
//
#include <hip/hip_runtime.h>
#include <hip/hip_bf16.h>

#define T_STEPS 36
#define P_DIM   64
#define H_DIM   128
#define W_DIM   256
#define HW      (H_DIM * W_DIM)       // 32768
#define PHW     (P_DIM * HW)          // 2097152
#define OUTY    (T_STEPS * HW)        // 1179648

#define LS     168                    // LDS row stride in halves (336B, 16B-aligned, bank-friendly)
#define LROWS  20                     // 16 rows + 2 halo each side
#define CH     (LROWS * LS)           // halves per channel plane

typedef _Float16 v16h __attribute__((ext_vector_type(16)));
typedef _Float16 v8h  __attribute__((ext_vector_type(8)));
typedef float    v8f  __attribute__((ext_vector_type(8)));

union Frag { v16h v; v8h h8[2]; };

// ---------------------------------------------------------------------------
// Init: zero the h state buffer
// ---------------------------------------------------------------------------
__global__ __launch_bounds__(256) void kZeroH(float* __restrict__ h) {
    size_t i = (size_t)blockIdx.x * 256 + threadIdx.x;
    if (i < PHW) h[i] = 0.0f;
}

// ---------------------------------------------------------------------------
// Precompute Toeplitz B fragments for all 3 convs in the WMMA B-layout:
// Btab[conv][c][dy][n][k] (f16), k in 0..31, n in 0..15.
// B[k,n] = w[c,dy,k-n] when 0 <= k-n <= 4, else 0.
// ---------------------------------------------------------------------------
__global__ __launch_bounds__(256) void kBtab(const float* __restrict__ wu,
                                             const float* __restrict__ wr,
                                             const float* __restrict__ wo,
                                             _Float16* __restrict__ Btab) {
    int idx = blockIdx.x * 256 + threadIdx.x;
    if (idx >= 3 * 2 * 5 * 16 * 32) return;
    int conv = idx / 5120;
    int r    = idx % 5120;
    int c    = r / 2560;  r %= 2560;
    int dy   = r / 512;   r %= 512;
    int n    = r / 32;
    int k    = r % 32;
    int dx   = k - n;
    const float* w = (conv == 0) ? wu : (conv == 1) ? wr : wo;
    float v = (dx >= 0 && dx <= 4) ? w[c * 25 + dy * 5 + dx] : 0.0f;
    Btab[idx] = (_Float16)v;
}

// ---------------------------------------------------------------------------
// Shared: load the 2-channel (16+4)x(128+16) f16 tile into LDS.
// ch0 = gam*x_t, ch1 = other (h or g). Zero pad outside the image (SAME conv).
// ---------------------------------------------------------------------------
__device__ __forceinline__ void load_tile(_Float16* __restrict__ smem,
                                          const float* __restrict__ xb, float gamp,
                                          const float* __restrict__ ob,
                                          int y0, int x0, int tid) {
    for (int i = tid; i < LROWS * 144; i += 256) {
        const int r  = i / 144;
        const int cc = i - r * 144;
        const int y  = y0 - 2 + r;
        const int x  = x0 - 2 + cc;
        float xv = 0.0f, ov = 0.0f;
        if ((unsigned)y < (unsigned)H_DIM && (unsigned)x < (unsigned)W_DIM) {
            xv = gamp * xb[y * W_DIM + x];
            ov = ob[y * W_DIM + x];
        }
        smem[r * LS + cc]      = (_Float16)xv;
        smem[CH + r * LS + cc] = (_Float16)ov;
    }
}

// ---------------------------------------------------------------------------
// Kernel A: u = sigmoid(conv([x,h], w_update)+bu); r = sigmoid(conv(.,w_reset)+br)
// Writes u and g = r*h to workspace.
// ---------------------------------------------------------------------------
__global__ __launch_bounds__(256) void kGates(
    const float* __restrict__ x_all, const float* __restrict__ h_buf,
    const float* __restrict__ gam,
    const float* __restrict__ b_upd, const float* __restrict__ b_rst,
    const _Float16* __restrict__ Btab,
    float* __restrict__ u_buf, float* __restrict__ g_buf, int t)
{
    __shared__ _Float16 smem[2 * CH];
    const int p  = blockIdx.z;
    const int y0 = blockIdx.y << 4;
    const int x0 = blockIdx.x << 7;
    const float gamp = gam[p];
    const float bu_s = b_upd[0];
    const float br_s = b_rst[0];
    const float* __restrict__ xb = x_all + ((size_t)t * P_DIM + p) * HW;
    const float* __restrict__ hb = h_buf + (size_t)p * HW;
    const int tid = threadIdx.x;

    load_tile(smem, xb, gamp, hb, y0, x0, tid);
    __syncthreads();

    const int lane = tid & 31, wave = tid >> 5;
    const int mn = lane & 15, laneHi = lane >> 4;   // A: m=lane%16; B/C/D: n=lane%16
    const int xw = wave << 4;

    v8f acc_u = {0.f,0.f,0.f,0.f,0.f,0.f,0.f,0.f};
    v8f acc_r = {0.f,0.f,0.f,0.f,0.f,0.f,0.f,0.f};

    #pragma unroll
    for (int dy = 0; dy < 5; ++dy) {
        #pragma unroll
        for (int c = 0; c < 2; ++c) {
            Frag a, bu, br;
            const _Float16* ap = smem + c * CH + (mn + dy) * LS + xw + laneHi * 8;
            a.h8[0] = *(const v8h*)(ap);
            a.h8[1] = *(const v8h*)(ap + 16);
            const _Float16* bup = Btab + (((0 * 2 + c) * 5 + dy) << 9) + (mn << 5) + laneHi * 8;
            const _Float16* brp = Btab + (((1 * 2 + c) * 5 + dy) << 9) + (mn << 5) + laneHi * 8;
            bu.h8[0] = *(const v8h*)(bup); bu.h8[1] = *(const v8h*)(bup + 16);
            br.h8[0] = *(const v8h*)(brp); br.h8[1] = *(const v8h*)(brp + 16);
            acc_u = __builtin_amdgcn_wmma_f32_16x16x32_f16(false, a.v, false, bu.v,
                                                           (short)0, acc_u, false, false);
            acc_r = __builtin_amdgcn_wmma_f32_16x16x32_f16(false, a.v, false, br.v,
                                                           (short)0, acc_r, false, false);
        }
    }

    const int xg = x0 + xw + mn;
    #pragma unroll
    for (int j = 0; j < 8; ++j) {
        const int yr = y0 + j + laneHi * 8;           // C/D layout: VGPR j -> M=j (+8 hi half)
        const size_t o = (size_t)p * HW + (size_t)yr * W_DIM + xg;
        const float uu = 1.0f / (1.0f + __expf(-(acc_u[j] + bu_s)));
        const float rr = 1.0f / (1.0f + __expf(-(acc_r[j] + br_s)));
        u_buf[o] = uu;
        g_buf[o] = rr * hb[(size_t)yr * W_DIM + xg];
    }
}

// ---------------------------------------------------------------------------
// Kernel B: c = conv([x, g], w_out); h = h*(1-u)+c*u (in place, pointwise-safe);
// xs = tanh(h)*add_weights -> d_out.
// ---------------------------------------------------------------------------
__global__ __launch_bounds__(256) void kOut(
    const float* __restrict__ x_all, float* __restrict__ h_buf,
    const float* __restrict__ gam, const _Float16* __restrict__ Btab,
    const float* __restrict__ u_buf, const float* __restrict__ g_buf,
    const float* __restrict__ aw, float* __restrict__ xs_out, int t)
{
    __shared__ _Float16 smem[2 * CH];
    const int p  = blockIdx.z;
    const int y0 = blockIdx.y << 4;
    const int x0 = blockIdx.x << 7;
    const float gamp = gam[p];
    const float* __restrict__ xb = x_all + ((size_t)t * P_DIM + p) * HW;
    const float* __restrict__ gb = g_buf + (size_t)p * HW;
    const int tid = threadIdx.x;

    load_tile(smem, xb, gamp, gb, y0, x0, tid);
    __syncthreads();

    const int lane = tid & 31, wave = tid >> 5;
    const int mn = lane & 15, laneHi = lane >> 4;
    const int xw = wave << 4;

    v8f acc = {0.f,0.f,0.f,0.f,0.f,0.f,0.f,0.f};
    #pragma unroll
    for (int dy = 0; dy < 5; ++dy) {
        #pragma unroll
        for (int c = 0; c < 2; ++c) {
            Frag a, b;
            const _Float16* ap = smem + c * CH + (mn + dy) * LS + xw + laneHi * 8;
            a.h8[0] = *(const v8h*)(ap);
            a.h8[1] = *(const v8h*)(ap + 16);
            const _Float16* bp = Btab + (((2 * 2 + c) * 5 + dy) << 9) + (mn << 5) + laneHi * 8;
            b.h8[0] = *(const v8h*)(bp); b.h8[1] = *(const v8h*)(bp + 16);
            acc = __builtin_amdgcn_wmma_f32_16x16x32_f16(false, a.v, false, b.v,
                                                         (short)0, acc, false, false);
        }
    }

    const int xg = x0 + xw + mn;
    #pragma unroll
    for (int j = 0; j < 8; ++j) {
        const int yr = y0 + j + laneHi * 8;
        const size_t o = (size_t)p * HW + (size_t)yr * W_DIM + xg;
        const float hv = h_buf[o];
        const float uu = u_buf[o];
        const float hn = hv * (1.0f - uu) + acc[j] * uu;
        h_buf[o] = hn;
        xs_out[((size_t)t * P_DIM + p) * HW + (size_t)yr * W_DIM + xg] = tanhf(hn) * aw[o];
    }
}

// ---------------------------------------------------------------------------
// out_y[t,y,x] = sum_p xs[t,p,y,x] + bias[y,x]  (deterministic serial-p sum)
// ---------------------------------------------------------------------------
__global__ __launch_bounds__(256) void kReduce(const float* __restrict__ xs,
                                               const float* __restrict__ bias,
                                               float* __restrict__ out_y) {
    const int i = blockIdx.x * 256 + threadIdx.x;
    if (i >= OUTY) return;
    const int t  = i / HW;
    const int yx = i - t * HW;
    float s = bias[yx];
    const float* b = xs + (size_t)t * PHW + yx;
    #pragma unroll 4
    for (int p = 0; p < P_DIM; ++p) s += b[(size_t)p * HW];
    out_y[i] = s;
}

// ---------------------------------------------------------------------------
extern "C" void kernel_launch(void* const* d_in, const int* in_sizes, int n_in,
                              void* d_out, int out_size, void* d_ws, size_t ws_size,
                              hipStream_t stream) {
    const float* inputs      = (const float*)d_in[0];
    const float* w_reset     = (const float*)d_in[1];
    const float* b_reset     = (const float*)d_in[2];
    const float* w_update    = (const float*)d_in[3];
    const float* b_update    = (const float*)d_in[4];
    const float* w_out       = (const float*)d_in[5];
    const float* gam         = (const float*)d_in[6];
    const float* add_weights = (const float*)d_in[7];
    const float* bias        = (const float*)d_in[8];
    float* out = (float*)d_out;

    float*    h_buf = (float*)d_ws;
    float*    u_buf = h_buf + PHW;
    float*    g_buf = u_buf + PHW;
    _Float16* Btab  = (_Float16*)(g_buf + PHW);   // 15360 halves

    kZeroH<<<PHW / 256, 256, 0, stream>>>(h_buf);
    kBtab<<<60, 256, 0, stream>>>(w_update, w_reset, w_out, Btab);

    dim3 grid(W_DIM / 128, H_DIM / 16, P_DIM);    // (2, 8, 64)
    for (int t = 0; t < T_STEPS; ++t) {
        kGates<<<grid, 256, 0, stream>>>(inputs, h_buf, gam, b_update, b_reset,
                                         Btab, u_buf, g_buf, t);
        kOut<<<grid, 256, 0, stream>>>(inputs, h_buf, gam, Btab,
                                       u_buf, g_buf, add_weights, out + OUTY, t);
    }
    kReduce<<<OUTY / 256, 256, 0, stream>>>(out + OUTY, bias, out);
}
